// DecoderRNN_1949915153058
// MI455X (gfx1250) — matrix operand order repaired and checked by reading, not compiled, not run on hardware
//
#include <hip/hip_runtime.h>
#include <hip/hip_bf16.h>

// ---------------------------------------------------------------------------
// DecoderRNN (2-layer LSTM + attention caption decoder) for gfx1250 / MI455X.
// Bandwidth-bound: weights downconverted once to bf16 and *fused along K*
// ([W_ih ‖ W_hh], [Wc ‖ Wh]) so every GEMM is a single branch-free
// v_wmma_f32_16x16x32_bf16 stream with software-pipelined b128 loads.
// wave32 fragment layouts: A/B in contiguous k order (k = 16*(lane/16) + s), D per CDNA5 ISA 7.12.2.
// ---------------------------------------------------------------------------

static constexpr int B_ = 32;
static constexpr int L_ = 512;
static constexpr int D_ = 2048;
static constexpr int T_ = 20;
static constexpr int V_ = 10000;
static constexpr int G_ = 4 * D_;   // 8192 gate width
static constexpr int K2_ = 2 * D_;  // 4096 fused-K

typedef __attribute__((ext_vector_type(16))) __bf16 v16bf;
typedef __attribute__((ext_vector_type(8)))  __bf16 v8bf;
typedef __attribute__((ext_vector_type(8)))  float  v8f;

__device__ __forceinline__ float sigmoidf_(float x) {
    return 1.0f / (1.0f + __expf(-x));
}

// ---------------------------------------------------------------------------
// fp32 -> bf16 (plain) and fused [Wa ‖ Wb] row-concat conversion
// ---------------------------------------------------------------------------
__global__ void f32_to_bf16_kernel(const float* __restrict__ src,
                                   __hip_bfloat16* __restrict__ dst, size_t n) {
    size_t i = (size_t)blockIdx.x * blockDim.x + threadIdx.x;
    size_t stride = (size_t)gridDim.x * blockDim.x;
    for (; i < n; i += stride) dst[i] = __float2bfloat16(src[i]);
}

__global__ void fuse_weights_bf16_kernel(const float* __restrict__ Wa,
                                         const float* __restrict__ Wb,
                                         __hip_bfloat16* __restrict__ dst,
                                         int Nrows, int Kc) {
    size_t t = (size_t)blockIdx.x * blockDim.x + threadIdx.x;
    size_t total = (size_t)Nrows * 2 * Kc;
    size_t stride = (size_t)gridDim.x * blockDim.x;
    for (; t < total; t += stride) {
        size_t n = t / (size_t)(2 * Kc);
        int k = (int)(t - n * (size_t)(2 * Kc));
        float v = (k < Kc) ? Wa[n * Kc + k] : Wb[n * Kc + (k - Kc)];
        dst[t] = __float2bfloat16(v);
    }
}

// ---------------------------------------------------------------------------
// h1 = c1 = mean over L of features (B,L,D); bf16 copies into fused buffers
// ---------------------------------------------------------------------------
__global__ void mean_features_kernel(const float* __restrict__ features,
                                     float* __restrict__ h1, float* __restrict__ c1,
                                     __hip_bfloat16* __restrict__ hb1, int ld1,
                                     __hip_bfloat16* __restrict__ hb2, int ld2) {
    int t = blockIdx.x * 256 + threadIdx.x;      // B*D threads
    int b = t / D_, d = t - b * D_;
    const float* f = features + (size_t)b * L_ * D_ + d;
    float s = 0.0f;
    for (int l = 0; l < L_; ++l) s += f[(size_t)l * D_];
    s *= (1.0f / (float)L_);
    h1[t] = s; c1[t] = s;
    __hip_bfloat16 bs = __float2bfloat16(s);
    hb1[(size_t)b * ld1 + d] = bs;
    hb2[(size_t)b * ld2 + d] = bs;
}

__global__ void init_layer2_kernel(const float* __restrict__ h2i,
                                   const float* __restrict__ c2i,
                                   float* __restrict__ h2, float* __restrict__ c2,
                                   __hip_bfloat16* __restrict__ hb1, int ld1,
                                   __hip_bfloat16* __restrict__ hb2, int ld2) {
    int t = blockIdx.x * 256 + threadIdx.x;      // B*D threads
    int b = t / D_, d = t - b * D_;
    float h = h2i[t];
    h2[t] = h; c2[t] = c2i[t];
    __hip_bfloat16 bh16 = __float2bfloat16(h);
    hb1[(size_t)b * ld1 + d] = bh16;
    hb2[(size_t)b * ld2 + d] = bh16;
}

// ---------------------------------------------------------------------------
// x = embed[captions[:,0]] into fused [x ‖ h1] buffer
// ---------------------------------------------------------------------------
__global__ void embed_caption_kernel(const int* __restrict__ captions,
                                     const __hip_bfloat16* __restrict__ embB,
                                     __hip_bfloat16* __restrict__ xh, int ld) {
    int d = blockIdx.x * 256 + threadIdx.x;
    int b = blockIdx.y;
    int cap = captions[(size_t)b * T_];          // captions[b, 0]
    xh[(size_t)b * ld + d] = embB[(size_t)cap * D_ + d];
}

// ---------------------------------------------------------------------------
// argmax over previous logits row (first-match tie-break, like jnp.argmax)
// ---------------------------------------------------------------------------
__global__ __launch_bounds__(256)
void argmax_kernel(const float* __restrict__ logits, int ld, int* __restrict__ idx) {
    int b = blockIdx.x, tid = threadIdx.x;
    const float* row = logits + (size_t)b * ld;
    __shared__ float sv[256];
    __shared__ int   si[256];
    float best = -3.4e38f; int bi = 0x7fffffff;
    for (int n = tid; n < V_; n += 256) {
        float v = row[n];
        if (v > best || (v == best && n < bi)) { best = v; bi = n; }
    }
    sv[tid] = best; si[tid] = bi; __syncthreads();
    for (int s = 128; s > 0; s >>= 1) {
        if (tid < s) {
            if (sv[tid + s] > sv[tid] ||
                (sv[tid + s] == sv[tid] && si[tid + s] < si[tid])) {
                sv[tid] = sv[tid + s]; si[tid] = si[tid + s];
            }
        }
        __syncthreads();
    }
    if (tid == 0) idx[b] = si[0];
}

__global__ void embed_gather_kernel(const int* __restrict__ idx,
                                    const __hip_bfloat16* __restrict__ embB,
                                    const float* __restrict__ embF32,
                                    __hip_bfloat16* __restrict__ xh, int ld,
                                    float* __restrict__ embRow) {
    int d = blockIdx.x * 256 + threadIdx.x;
    int b = blockIdx.y;
    int id = idx[b];
    xh[(size_t)b * ld + d]     = embB[(size_t)id * D_ + d];
    embRow[(size_t)b * D_ + d] = embF32[(size_t)id * D_ + d];
}

// ---------------------------------------------------------------------------
// Elementwise LSTM cell: gates (B,4D) -> h,c fp32 + two bf16 h copies
// (written into the fused [·‖h] activation buffers for the next GEMMs)
// ---------------------------------------------------------------------------
__global__ void lstm_cell_kernel(const float* __restrict__ gates,
                                 float* __restrict__ h, float* __restrict__ c,
                                 __hip_bfloat16* __restrict__ hb1, int ld1,
                                 __hip_bfloat16* __restrict__ hb2, int ld2) {
    int t = blockIdx.x * 256 + threadIdx.x;      // B*D threads
    int b = t / D_, d = t - b * D_;
    const float* gr = gates + (size_t)b * G_;
    float ig = sigmoidf_(gr[d]);
    float fg = sigmoidf_(gr[D_ + d]);
    float gg = tanhf(gr[2 * D_ + d]);
    float og = sigmoidf_(gr[3 * D_ + d]);
    float cn = fg * c[t] + ig * gg;
    float hn = og * tanhf(cn);
    c[t] = cn; h[t] = hn;
    __hip_bfloat16 bh16 = __float2bfloat16(hn);
    hb1[(size_t)b * ld1 + d] = bh16;
    hb2[(size_t)b * ld2 + d] = bh16;
}

// ---------------------------------------------------------------------------
// Attention: scores = h2 . features[b,l,:], softmax over L, context -> bf16
// One block (256 threads) per batch element, fp32 throughout.
// ---------------------------------------------------------------------------
__global__ __launch_bounds__(256)
void attention_kernel(const float* __restrict__ h2,
                      const float* __restrict__ features,
                      __hip_bfloat16* __restrict__ ctxB, int ldCtx) {
    const int b = blockIdx.x, tid = threadIdx.x;
    __shared__ float sc[L_];
    __shared__ float red[256];
    const float* fb = features + (size_t)b * L_ * D_;
    const float* hb = h2 + (size_t)b * D_;
    for (int l = tid; l < L_; l += 256) {
        const float* fr = fb + (size_t)l * D_;
        float s = 0.0f;
        for (int d = 0; d < D_; ++d) s += hb[d] * fr[d];
        sc[l] = s;
    }
    __syncthreads();
    float m = -3.4e38f;
    for (int l = tid; l < L_; l += 256) m = fmaxf(m, sc[l]);
    red[tid] = m; __syncthreads();
    for (int s = 128; s > 0; s >>= 1) {
        if (tid < s) red[tid] = fmaxf(red[tid], red[tid + s]);
        __syncthreads();
    }
    m = red[0]; __syncthreads();
    float ps = 0.0f;
    for (int l = tid; l < L_; l += 256) { float e = __expf(sc[l] - m); sc[l] = e; ps += e; }
    red[tid] = ps; __syncthreads();
    for (int s = 128; s > 0; s >>= 1) {
        if (tid < s) red[tid] += red[tid + s];
        __syncthreads();
    }
    float inv = 1.0f / red[0];
    for (int d = tid; d < D_; d += 256) {
        float a = 0.0f;
        for (int l = 0; l < L_; ++l) a += sc[l] * fb[(size_t)l * D_ + d];
        ctxB[(size_t)b * ldCtx + d] = __float2bfloat16(a * inv);
    }
}

// ---------------------------------------------------------------------------
// WMMA GEMM:  out[m,n] = sum_k A[m,k]*W[n,k] + bias1[n] + bias2[n] + addend
// A: (32,K-cols) bf16, row stride lda.  W: (N,K) bf16 row-major (== B^T, so
// each lane's B-fragment is a contiguous 32-byte read of one W row).
// One wave owns a 16x16 C tile; 8 waves/block; K unrolled by 64 with two
// independent fragment sets so loads pipeline under the WMMAs.
// K must be a multiple of 64 (4096 / 2048 here).
// ---------------------------------------------------------------------------
__global__ __launch_bounds__(256)
void gemm_bf16_wmma(const __hip_bfloat16* __restrict__ Ap, int lda,
                    const __hip_bfloat16* __restrict__ Wp,
                    const float* __restrict__ bias1,
                    const float* __restrict__ bias2,
                    const float* __restrict__ addend, int ldAdd,
                    float* __restrict__ outF, int ldOutF,
                    __hip_bfloat16* __restrict__ outB, int ldOutB,
                    int N, int K) {
    const int lane = threadIdx.x & 31;
    const int wid  = threadIdx.x >> 5;
    const int nTiles = N >> 4;
    const int totalTiles = nTiles * 2;           // M = 32 -> 2 tile rows
    const int tile = blockIdx.x * 8 + wid;
    if (tile >= totalTiles) return;              // wave-uniform exit
    const int tm = tile / nTiles;
    const int tn = tile - tm * nTiles;
    const int m0 = tm * 16, n0 = tn * 16;
    const int g  = lane >> 4;                    // half-wave group
    const int ln = lane & 15;

    const __bf16* A = reinterpret_cast<const __bf16*>(Ap);
    const __bf16* W = reinterpret_cast<const __bf16*>(Wp);

    // A fragment: lane holds M=ln, K = k0+16g..+16 (contiguous)
    const __bf16* a = A + (size_t)(m0 + ln) * lda + 16 * g;
    // B fragment: lane holds N=ln, K = k0+16g..+16 (contiguous row of W)
    const __bf16* w = W + (size_t)(n0 + ln) * K + 16 * g;

    v8f acc = {};
    union Frag { v16bf v; v8bf h[2]; };

    for (int k0 = 0; k0 < K; k0 += 64) {
        // issue all 8 b128 loads up front; two independent fragment sets
        v8bf a0lo = *(const v8bf*)(a + k0);
        v8bf a0hi = *(const v8bf*)(a + k0 + 8);
        v16bf b0  = *(const v16bf*)(w + k0);
        v8bf a1lo = *(const v8bf*)(a + k0 + 32);
        v8bf a1hi = *(const v8bf*)(a + k0 + 40);
        v16bf b1  = *(const v16bf*)(w + k0 + 32);
        __builtin_prefetch((const void*)(w + k0 + 1024), 0, 3);
        Frag f0; f0.h[0] = a0lo; f0.h[1] = a0hi;
        acc = __builtin_amdgcn_wmma_f32_16x16x32_bf16(
                  false, f0.v, false, b0, (short)0, acc, false, false);
        Frag f1; f1.h[0] = a1lo; f1.h[1] = a1hi;
        acc = __builtin_amdgcn_wmma_f32_16x16x32_bf16(
                  false, f1.v, false, b1, (short)0, acc, false, false);
    }

    const int n = n0 + ln;
    float bias = 0.0f;
    if (bias1) bias += bias1[n];
    if (bias2) bias += bias2[n];
#pragma unroll
    for (int r = 0; r < 8; ++r) {                // D layout: M = r + 8g, N = ln
        const int m = m0 + r + 8 * g;
        float v = acc[r] + bias;
        if (addend) v += addend[(size_t)m * ldAdd + n];
        if (outF) outF[(size_t)m * ldOutF + n] = v;
        if (outB) outB[(size_t)m * ldOutB + n] = __float2bfloat16(v);
    }
}

// ---------------------------------------------------------------------------
// Host orchestration
// ---------------------------------------------------------------------------
extern "C" void kernel_launch(void* const* d_in, const int* in_sizes, int n_in,
                              void* d_out, int out_size, void* d_ws, size_t ws_size,
                              hipStream_t stream) {
    const float* features = (const float*)d_in[0];
    const int*   captions = (const int*)d_in[1];
    const float* h2_init  = (const float*)d_in[2];
    const float* c2_init  = (const float*)d_in[3];
    const float* embed    = (const float*)d_in[4];
    const float* W_ih1 = (const float*)d_in[5];
    const float* W_hh1 = (const float*)d_in[6];
    const float* b_ih1 = (const float*)d_in[7];
    const float* b_hh1 = (const float*)d_in[8];
    const float* W_ih2 = (const float*)d_in[9];
    const float* W_hh2 = (const float*)d_in[10];
    const float* b_ih2 = (const float*)d_in[11];
    const float* b_hh2 = (const float*)d_in[12];
    const float* Wc = (const float*)d_in[13];
    const float* bc = (const float*)d_in[14];
    const float* Wh = (const float*)d_in[15];
    const float* bh = (const float*)d_in[16];
    const float* Wo = (const float*)d_in[17];
    const float* bo = (const float*)d_in[18];
    float* out = (float*)d_out;                  // (B, T, V)

    // ---- workspace carve ----
    char* p = (char*)d_ws;
    auto carve = [&](size_t bytes) {
        char* r = p; p += (bytes + 255) & ~(size_t)255; return (void*)r;
    };
    // fused weights: rows are [W_a row ‖ W_b row], length 2D
    __hip_bfloat16* W1f   = (__hip_bfloat16*)carve((size_t)G_ * K2_ * 2);
    __hip_bfloat16* W2f   = (__hip_bfloat16*)carve((size_t)G_ * K2_ * 2);
    __hip_bfloat16* WcWhf = (__hip_bfloat16*)carve((size_t)D_ * K2_ * 2);
    __hip_bfloat16* Wob   = (__hip_bfloat16*)carve((size_t)V_ * D_ * 2);
    __hip_bfloat16* embB  = (__hip_bfloat16*)carve((size_t)V_ * D_ * 2);
    float* h1f   = (float*)carve((size_t)B_ * D_ * 4);
    float* c1f   = (float*)carve((size_t)B_ * D_ * 4);
    float* h2f   = (float*)carve((size_t)B_ * D_ * 4);
    float* c2f   = (float*)carve((size_t)B_ * D_ * 4);
    float* gates = (float*)carve((size_t)B_ * G_ * 4);
    float* embRow= (float*)carve((size_t)B_ * D_ * 4);
    // fused activation buffers, rows of length 2D
    __hip_bfloat16* xh1  = (__hip_bfloat16*)carve((size_t)B_ * K2_ * 2); // [x  ‖ h1]
    __hip_bfloat16* xh2  = (__hip_bfloat16*)carve((size_t)B_ * K2_ * 2); // [h1 ‖ h2]
    __hip_bfloat16* ctxh = (__hip_bfloat16*)carve((size_t)B_ * K2_ * 2); // [ctx‖ h2]
    __hip_bfloat16* Sb   = (__hip_bfloat16*)carve((size_t)B_ * D_ * 2);
    int* idx = (int*)carve((size_t)B_ * 4);

    auto conv = [&](const float* s, __hip_bfloat16* dst, size_t n) {
        int blocks = (int)((n + 255) / 256);
        if (blocks > 8192) blocks = 8192;
        f32_to_bf16_kernel<<<blocks, 256, 0, stream>>>(s, dst, n);
    };
    auto fuse = [&](const float* Wa, const float* Wb, __hip_bfloat16* dst, int N) {
        size_t n = (size_t)N * K2_;
        int blocks = (int)((n + 255) / 256);
        if (blocks > 8192) blocks = 8192;
        fuse_weights_bf16_kernel<<<blocks, 256, 0, stream>>>(Wa, Wb, dst, N, D_);
    };
    auto gemm = [&](const __hip_bfloat16* A, int lda, const __hip_bfloat16* W,
                    const float* bias1, const float* bias2,
                    const float* addend, int ldAdd,
                    float* oF, int ldF, __hip_bfloat16* oB, int ldB,
                    int N, int K) {
        int tiles = (N / 16) * 2;
        int blocks = (tiles + 7) / 8;
        gemm_bf16_wmma<<<blocks, 256, 0, stream>>>(A, lda, W, bias1, bias2,
                                                   addend, ldAdd, oF, ldF, oB, ldB,
                                                   N, K);
    };

    // ---- one-time (per call) bf16 weight conversion, K-fused ----
    fuse(W_ih1, W_hh1, W1f, G_);
    fuse(W_ih2, W_hh2, W2f, G_);
    fuse(Wc,    Wh,    WcWhf, D_);
    conv(Wo,    Wob,  (size_t)V_ * D_);
    conv(embed, embB, (size_t)V_ * D_);

    const int bdBlocks = (B_ * D_) / 256;        // 256
    // h1 bf16 -> xh1[:,D:2D] and xh2[:,0:D]
    mean_features_kernel<<<bdBlocks, 256, 0, stream>>>(features, h1f, c1f,
                                                       xh1 + D_, K2_, xh2, K2_);
    // h2 bf16 -> xh2[:,D:2D] and ctxh[:,D:2D]
    init_layer2_kernel<<<bdBlocks, 256, 0, stream>>>(h2_init, c2_init, h2f, c2f,
                                                     xh2 + D_, K2_, ctxh + D_, K2_);

    // ---- t = 0 ----
    embed_caption_kernel<<<dim3(D_ / 256, B_), 256, 0, stream>>>(captions, embB,
                                                                 xh1, K2_);
    gemm(xh1, K2_, W1f, b_ih1, b_hh1, nullptr, 0,
         gates, G_, nullptr, 0, G_, K2_);
    lstm_cell_kernel<<<bdBlocks, 256, 0, stream>>>(gates, h1f, c1f,
                                                   xh1 + D_, K2_, xh2, K2_);
    gemm(xh2, K2_, W2f, b_ih2, b_hh2, nullptr, 0,
         gates, G_, nullptr, 0, G_, K2_);
    lstm_cell_kernel<<<bdBlocks, 256, 0, stream>>>(gates, h2f, c2f,
                                                   xh2 + D_, K2_, ctxh + D_, K2_);
    // out[:, 0, :] = h2 @ Wo^T + bo   (h2 bf16 read out of xh2[:,D:2D])
    gemm(xh2 + D_, K2_, Wob, bo, nullptr, nullptr, 0,
         out + 0, T_ * V_, nullptr, 0, V_, D_);

    // ---- t = 1 .. T-1 ----
    for (int t = 1; t < T_; ++t) {
        argmax_kernel<<<B_, 256, 0, stream>>>(out + (size_t)(t - 1) * V_, T_ * V_, idx);
        embed_gather_kernel<<<dim3(D_ / 256, B_), 256, 0, stream>>>(idx, embB, embed,
                                                                    xh1, K2_, embRow);
        gemm(xh1, K2_, W1f, b_ih1, b_hh1, nullptr, 0,
             gates, G_, nullptr, 0, G_, K2_);
        lstm_cell_kernel<<<bdBlocks, 256, 0, stream>>>(gates, h1f, c1f,
                                                       xh1 + D_, K2_, xh2, K2_);
        gemm(xh2, K2_, W2f, b_ih2, b_hh2, nullptr, 0,
             gates, G_, nullptr, 0, G_, K2_);
        lstm_cell_kernel<<<bdBlocks, 256, 0, stream>>>(gates, h2f, c2f,
                                                       xh2 + D_, K2_, ctxh + D_, K2_);
        attention_kernel<<<B_, 256, 0, stream>>>(h2f, features, ctxh, K2_);
        // S = ctx@Wc^T + h2@Wh^T + bc + bh + embed[idx]  (bf16 for next GEMM)
        gemm(ctxh, K2_, WcWhf, bc, bh, embRow, D_,
             nullptr, 0, Sb, D_, D_, K2_);
        gemm(Sb, D_, Wob, bo, nullptr, nullptr, 0,
             out + (size_t)t * V_, T_ * V_, nullptr, 0, V_, D_);
    }
}
